// PillarSeg_90950227460811
// MI455X (gfx1250) — matrix operand order repaired and checked by reading, not compiled, not run on hardware
//
#include <hip/hip_runtime.h>
#include <hip/hip_bf16.h>

typedef __attribute__((ext_vector_type(16))) _Float16 v16h;
typedef __attribute__((ext_vector_type(8)))  float    v8f;

#define NPTS   200000
#define BATCH  4
#define TOTAL  (BATCH*NPTS)
#define HDIM   512
#define WDIM   512
#define CDIM   64
#define PCNT   150000
#define RCNT   100000
#define WPB    4            // waves per block (block = 128 threads)

// weight-fragment chunk bases in d_ws (each chunk = 8 vgprs * 32 lanes * 4B = 1KB)
#define CH_WF  0            // W_flow: 2 ntiles * 3 kchunks
#define CH_WP  6            // W_pc:   2 * 3
#define CH_W1  12           // W1:     2 * 2
#define CH_W2  16           // W2:     1 * 1
#define NCHUNK 17

union FragU { v16h h; unsigned u[8]; };

#define WMMA_F16(a, b, c) \
  __builtin_amdgcn_wmma_f32_16x16x32_f16(false, (a), false, (b), (short)0, (c), false, false)

__device__ __forceinline__ float gelu_exact(float x) {
  return 0.5f * x * (1.0f + erff(x * 0.70710678118654752440f));
}
__device__ __forceinline__ float sigmoidf(float x) {
  return 1.0f / (1.0f + expf(-x));
}

// ---------------------------------------------------------------------------
// Pack fp32 weights into f16 WMMA B-operand fragments.
// B chunk layout (wave32, 32x16 f16 B for v_wmma_f32_16x16x32_f16):
//   lane l: N = n0 + (l%16); lanes 0-15 hold K = k0+0..15, lanes 16-31 hold
//   K = k0+16..31; VGPR v holds (K = base+2v, base+2v+1) packed in one dword.
// Stored as ws[chunk*256 + v*32 + lane].
// ---------------------------------------------------------------------------
__global__ __launch_bounds__(256) void pack_weights_kernel(
    const float* __restrict__ Wf, const float* __restrict__ Wp,
    const float* __restrict__ W1, const float* __restrict__ W2,
    unsigned* __restrict__ ws)
{
  const int e = threadIdx.x;          // 0..255
  const int v = e >> 5;
  const int l = e & 31;
  for (int c = 0; c < NCHUNK; ++c) {
    const float* src; int K, Nfan, nt, kc;
    if (c < 6)        { src = Wf; K = 70; Nfan = 32; nt = c / 3;        kc = c % 3;        }
    else if (c < 12)  { src = Wp; K = 96; Nfan = 32; nt = (c - 6) / 3;  kc = (c - 6) % 3;  }
    else if (c < 16)  { src = W1; K = 64; Nfan = 32; nt = (c - 12) / 2; kc = (c - 12) % 2; }
    else              { src = W2; K = 32; Nfan = 16; nt = 0;            kc = 0;            }
    const int n  = nt * 16 + (l & 15);
    const int kb = kc * 32 + ((l >= 16) ? 16 : 0) + 2 * v;
    _Float16 h0 = (kb     < K) ? (_Float16)src[(kb    ) * Nfan + n] : (_Float16)0.0f;
    _Float16 h1 = (kb + 1 < K) ? (_Float16)src[(kb + 1) * Nfan + n] : (_Float16)0.0f;
    union { _Float16 h[2]; unsigned u; } pk;
    pk.h[0] = h0; pk.h[1] = h1;
    ws[c * 256 + e] = pk.u;
  }
}

// Load an A fragment (16x32 f16) from an LDS row base, in contiguous k order (same as the B packing):
// lane l: M = l%16; half = l/16; VGPR v: K = 16*half + 2*v.
__device__ __forceinline__ v16h load_a_frag(const _Float16* row_base, int k0, int half) {
  FragU f;
#pragma unroll
  for (int v = 0; v < 8; ++v) {
    const int k = k0 + (half ? 16 : 0) + 2 * v;
    f.u[v] = *(const unsigned*)(row_base + k);   // ds_load_b32 (k even -> 4B aligned)
  }
  return f.h;
}

__device__ __forceinline__ v16h load_b_frag(const unsigned* __restrict__ ws, int chunk, int lane) {
  FragU f;
#pragma unroll
  for (int v = 0; v < 8; ++v) f.u[v] = ws[chunk * 256 + v * 32 + lane];
  return f.h;
}

// ---------------------------------------------------------------------------
// Main kernel: one wave32 processes a 16-point tile through the whole MLP.
// ---------------------------------------------------------------------------
__global__ __launch_bounds__(WPB * 32) void pillarseg_kernel(
    const float* __restrict__ pc0_map,   const float* __restrict__ flow_map,
    const float* __restrict__ pc0_fea,   const float* __restrict__ flows,
    const float* __restrict__ pose_flows,const float* __restrict__ radar_pose,
    const int*   __restrict__ voxel,     const int*   __restrict__ pidx,
    const float* __restrict__ b_flow,    const float* __restrict__ b_pc,
    const float* __restrict__ b1,        const float* __restrict__ b2,
    const float* __restrict__ W3,        const float* __restrict__ b3,
    const unsigned* __restrict__ ws,     float* __restrict__ out)
{
  __shared__ _Float16 sAf[WPB][16][96];   // concat_flow tile (f16, K-padded)
  __shared__ _Float16 sAp[WPB][16][96];   // concat_pc tile
  __shared__ _Float16 sH [WPB][16][64];   // layer-0 output (flow_feat|pc_feat)
  __shared__ _Float16 sH2[WPB][16][32];   // layer-1 output (post-GELU)
  __shared__ float    sH3[WPB][16][16];   // layer-2 output (post-GELU, f32)

  const int lane = threadIdx.x & 31;
  const int wid  = threadIdx.x >> 5;
  const int tile = blockIdx.x * WPB + wid;
  const int r    = lane & 15;             // row within tile / N index of frags
  const int half = lane >> 4;

  const int  gp  = tile * 16 + r;
  const bool ok  = (gp < TOTAL);
  const int  gpc = ok ? gp : 0;
  const int  bb  = gpc / NPTS;
  const int  nn  = gpc - bb * NPTS;

  // --- gather stage: fill LDS A-tiles ------------------------------------
  const long vbase = ((long)bb * NPTS + nn) * 3;
  int y = voxel[vbase + 1] & (HDIM - 1);
  int x = voxel[vbase + 2] & (WDIM - 1);
  const long plane   = (long)HDIM * WDIM;
  const long mapbase = ((long)bb * CDIM + half * 32) * plane + (long)y * WDIM + x;
  const float* fm = flow_map + mapbase;
  const float* pm = pc0_map  + mapbase;
#pragma unroll 8
  for (int c = 0; c < 32; ++c) {
    sAf[wid][r][half * 32 + c] = (_Float16)fm[(long)c * plane];
    sAp[wid][r][half * 32 + c] = (_Float16)pm[(long)c * plane];
  }
  if (half == 0) {
    sAf[wid][r][64] = (_Float16)flows[vbase + 0];
    sAf[wid][r][65] = (_Float16)flows[vbase + 1];
    sAf[wid][r][66] = (_Float16)flows[vbase + 2];
    const int pi = pidx[(long)bb * NPTS + nn];
    const float* ps = (pi < PCNT)
        ? (pose_flows + ((long)bb * PCNT + pi) * 3)
        : (radar_pose + ((long)bb * RCNT + (pi - PCNT)) * 3);
    sAf[wid][r][67] = (_Float16)ps[0];
    sAf[wid][r][68] = (_Float16)ps[1];
    sAf[wid][r][69] = (_Float16)ps[2];
#pragma unroll
    for (int j = 70; j < 96; ++j) sAf[wid][r][j] = (_Float16)0.0f;
  } else {
    const long fbase = ((long)bb * NPTS + nn) * 32;
#pragma unroll 8
    for (int j = 0; j < 32; ++j) sAp[wid][r][64 + j] = (_Float16)pc0_fea[fbase + j];
  }
  __syncthreads();

  // --- layer 0: flow_feat = Af(16x70)@Wf, pc_feat = Ap(16x96)@Wp ----------
  const v16h af0 = load_a_frag(&sAf[wid][r][0],  0, half);
  const v16h af1 = load_a_frag(&sAf[wid][r][0], 32, half);
  const v16h af2 = load_a_frag(&sAf[wid][r][0], 64, half);
  const v16h ap0 = load_a_frag(&sAp[wid][r][0],  0, half);
  const v16h ap1 = load_a_frag(&sAp[wid][r][0], 32, half);
  const v16h ap2 = load_a_frag(&sAp[wid][r][0], 64, half);

#pragma unroll
  for (int nt = 0; nt < 2; ++nt) {
    v8f cf = {};
    cf = WMMA_F16(af0, load_b_frag(ws, CH_WF + nt * 3 + 0, lane), cf);
    cf = WMMA_F16(af1, load_b_frag(ws, CH_WF + nt * 3 + 1, lane), cf);
    cf = WMMA_F16(af2, load_b_frag(ws, CH_WF + nt * 3 + 2, lane), cf);
    v8f cp = {};
    cp = WMMA_F16(ap0, load_b_frag(ws, CH_WP + nt * 3 + 0, lane), cp);
    cp = WMMA_F16(ap1, load_b_frag(ws, CH_WP + nt * 3 + 1, lane), cp);
    cp = WMMA_F16(ap2, load_b_frag(ws, CH_WP + nt * 3 + 2, lane), cp);
    const float bf = b_flow[nt * 16 + r];
    const float bp = b_pc  [nt * 16 + r];
    // D layout: element (M = v + 8*half, N = r); h = [flow_feat | pc_feat]
#pragma unroll
    for (int v = 0; v < 8; ++v) {
      sH[wid][v + 8 * half][     nt * 16 + r] = (_Float16)(cf[v] + bf);
      sH[wid][v + 8 * half][32 + nt * 16 + r] = (_Float16)(cp[v] + bp);
    }
  }
  __syncthreads();

  // --- layer 1: h(16x64) @ W1(64x32), GELU --------------------------------
  const v16h ah0 = load_a_frag(&sH[wid][r][0],  0, half);
  const v16h ah1 = load_a_frag(&sH[wid][r][0], 32, half);
#pragma unroll
  for (int nt = 0; nt < 2; ++nt) {
    v8f c = {};
    c = WMMA_F16(ah0, load_b_frag(ws, CH_W1 + nt * 2 + 0, lane), c);
    c = WMMA_F16(ah1, load_b_frag(ws, CH_W1 + nt * 2 + 1, lane), c);
    const float bias = b1[nt * 16 + r];
#pragma unroll
    for (int v = 0; v < 8; ++v)
      sH2[wid][v + 8 * half][nt * 16 + r] = (_Float16)gelu_exact(c[v] + bias);
  }
  __syncthreads();

  // --- layer 2: h2(16x32) @ W2(32x16), GELU -------------------------------
  const v16h a2 = load_a_frag(&sH2[wid][r][0], 0, half);
  v8f c2 = {};
  c2 = WMMA_F16(a2, load_b_frag(ws, CH_W2, lane), c2);
  const float bias2 = b2[r];
#pragma unroll
  for (int v = 0; v < 8; ++v)
    sH3[wid][v + 8 * half][r] = gelu_exact(c2[v] + bias2);
  __syncthreads();

  // --- layer 3: 16->1 dot + sigmoid; write mask + score -------------------
  if (half == 0) {
    float acc = b3[0];
#pragma unroll
    for (int k = 0; k < 16; ++k) acc += sH3[wid][r][k] * W3[k];
    const float score = sigmoidf(acc);
    if (ok) {
      out[gp]         = (score > 0.5f) ? 1.0f : 0.0f;  // mask
      out[TOTAL + gp] = score;                          // score
    }
  }
}

extern "C" void kernel_launch(void* const* d_in, const int* in_sizes, int n_in,
                              void* d_out, int out_size, void* d_ws, size_t ws_size,
                              hipStream_t stream) {
  const float* pc0_map    = (const float*)d_in[0];
  const float* flow_map   = (const float*)d_in[1];
  const float* pc0_fea    = (const float*)d_in[2];
  const float* flows      = (const float*)d_in[3];
  const float* pose_flows = (const float*)d_in[4];
  const float* radar_pose = (const float*)d_in[5];
  const int*   voxel      = (const int*)d_in[6];
  const int*   pidx       = (const int*)d_in[7];
  const float* W_flow     = (const float*)d_in[8];
  const float* b_flow     = (const float*)d_in[9];
  const float* W_pc       = (const float*)d_in[10];
  const float* b_pc       = (const float*)d_in[11];
  const float* W1         = (const float*)d_in[12];
  const float* b1         = (const float*)d_in[13];
  const float* W2         = (const float*)d_in[14];
  const float* b2         = (const float*)d_in[15];
  const float* W3         = (const float*)d_in[16];
  const float* b3         = (const float*)d_in[17];
  (void)in_sizes; (void)n_in; (void)out_size; (void)ws_size;

  unsigned* ws = (unsigned*)d_ws;

  pack_weights_kernel<<<1, 256, 0, stream>>>(W_flow, W_pc, W1, W2, ws);

  const int tiles  = (TOTAL + 15) / 16;            // 50000
  const int blocks = (tiles + WPB - 1) / WPB;      // 12500
  pillarseg_kernel<<<blocks, WPB * 32, 0, stream>>>(
      pc0_map, flow_map, pc0_fea, flows, pose_flows, radar_pose,
      voxel, pidx, b_flow, b_pc, b1, b2, W3, b3, ws, (float*)d_out);
}
